// KNNConvBlock_47820165874127
// MI455X (gfx1250) — matrix operand order repaired and checked by reading, not compiled, not run on hardware
//
#include <hip/hip_runtime.h>
#include <stdint.h>

typedef __attribute__((ext_vector_type(16))) _Float16 v16h;
typedef __attribute__((ext_vector_type(8)))  _Float16 v8h;
typedef __attribute__((ext_vector_type(8)))  float    v8f;

#define H_      64
#define W_      2048
#define HW_     (H_ * W_)
#define C_      5
#define STEM_   32
#define KNN_    9
#define FSTRIDE 72      // halves per feature row: 144B (16B multiple, odd 16B-unit stride)
#define OSTRIDE 260     // floats per staged row: 1040B (16B multiple, stride%64==4)

__device__ __forceinline__ float ldpix(const float* __restrict__ p, int b, int ch, int h, int w) {
    if ((unsigned)h < (unsigned)H_ && (unsigned)w < (unsigned)W_)
        return p[(((size_t)b * C_ + ch) * H_ + h) * W_ + w];
    return 0.0f;
}

// Order-preserving u32 key: |distance| bits with the 6-bit neighbor index in the
// LSBs (ties resolve to lower index, matching jax.lax.top_k ordering).
__device__ __forceinline__ uint32_t mkkey(float d, int i) {
    uint32_t bits = __float_as_uint(d);
    uint32_t ord  = ((int)bits < 0) ? ~bits : (bits | 0x80000000u);
    return (ord & 0xFFFFFFC0u) | (uint32_t)i;
}

__global__ void __launch_bounds__(256)
knn_conv_kernel(const float* __restrict__ x, const float* __restrict__ px,
                const float* __restrict__ wr, const float* __restrict__ wp,
                float* __restrict__ out)
{
    __shared__ _Float16 featA[256 * FSTRIDE];   // branch-1 features, f16, K padded to 64
    __shared__ _Float16 featB[256 * FSTRIDE];   // branch-2 features
    __shared__ float    ostage[STEM_ * OSTRIDE];

    const int tid  = threadIdx.x;
    const int lane = tid & 31;
    const int wave = tid >> 5;

    const int blocksPerBatch = HW_ / 256;                 // 512
    const int b      = blockIdx.x / blocksPerBatch;
    const int hwBase = (blockIdx.x % blocksPerBatch) * 256;
    const int hw = hwBase + tid;
    const int h  = hw >> 11;          // W_ = 2048
    const int w  = hw & (W_ - 1);

    // ---------------- phase 1: per-pixel KNN + feature build ----------------
    const float ctr = ldpix(x, b, 0, h, w);

    // branch 1: 5x5 search over x channel 0 (center forced smallest, diff = -1)
    uint32_t keys[KNN_];
    #pragma unroll
    for (int q = 0; q < KNN_; ++q) keys[q] = 0xFFFFFFFFu;
    #pragma unroll
    for (int i = 0; i < 25; ++i) {
        const int di = i / 5 - 2, dj = i % 5 - 2;
        float v = ldpix(x, b, 0, h + di, w + dj);
        float d = (i == 12) ? -1.0f : fabsf(v - ctr);
        uint32_t key = mkkey(d, i);
        #pragma unroll
        for (int q = 0; q < KNN_; ++q) {           // sorted insertion: v_min/v_max_u32
            uint32_t lo = key < keys[q] ? key : keys[q];
            uint32_t hi = key < keys[q] ? keys[q] : key;
            keys[q] = lo; key = hi;
        }
    }
    int knn[KNN_];
    #pragma unroll
    for (int q = 0; q < KNN_; ++q) knn[q] = (int)(keys[q] & 63u);

    // gather branch-1 features into packed v8h registers (K padding = zero init)
    v8h pkA[8];
    #pragma unroll
    for (int g = 0; g < 8; ++g) pkA[g] = (v8h)0;
    #pragma unroll
    for (int c = 0; c < C_; ++c) {
        #pragma unroll
        for (int j = 0; j < KNN_; ++j) {
            int i = knn[j];
            int di = i / 5 - 2, dj = i % 5 - 2;
            const int k = c * KNN_ + j;
            pkA[k >> 3][k & 7] = (_Float16)ldpix(x, b, c, h + di, w + dj);
        }
    }
    {
        _Float16* fa = &featA[tid * FSTRIDE];
        #pragma unroll
        for (int g = 0; g < 8; ++g) *(v8h*)&fa[g * 8] = pkA[g];   // 8x ds_store_b128
    }

    // branch 2: 7x7 search over pre_x channel 0, distance to x's center value
    #pragma unroll
    for (int q = 0; q < KNN_; ++q) keys[q] = 0xFFFFFFFFu;
    #pragma unroll
    for (int i = 0; i < 49; ++i) {
        const int di = i / 7 - 3, dj = i % 7 - 3;
        float v = ldpix(px, b, 0, h + di, w + dj);
        uint32_t key = mkkey(fabsf(v - ctr), i);
        #pragma unroll
        for (int q = 0; q < KNN_; ++q) {
            uint32_t lo = key < keys[q] ? key : keys[q];
            uint32_t hi = key < keys[q] ? keys[q] : key;
            keys[q] = lo; key = hi;
        }
    }
    int pknn[KNN_];
    #pragma unroll
    for (int q = 0; q < KNN_; ++q) pknn[q] = (int)(keys[q] & 63u);

    v8h pkB[8];
    #pragma unroll
    for (int g = 0; g < 8; ++g) pkB[g] = (v8h)0;
    float pc1[KNN_], pc2[KNN_], pc3[KNN_];
    #pragma unroll
    for (int j = 0; j < KNN_; ++j) {
        int i = pknn[j];
        int di = i / 7 - 3, dj = i % 7 - 3;
        pkB[j >> 3][j & 7] = (_Float16)ldpix(px, b, 0, h + di, w + dj);
        pc1[j] = ldpix(px, b, 1, h + di, w + dj);
        pc2[j] = ldpix(px, b, 2, h + di, w + dj);
        pc3[j] = ldpix(px, b, 3, h + di, w + dj);
        const int k4 = 36 + j;
        pkB[k4 >> 3][k4 & 7] = (_Float16)ldpix(px, b, 4, h + di, w + dj);
    }
    const float a1 = ldpix(x, b, 1, h, w);
    const float a2 = ldpix(x, b, 2, h, w);
    const float a3 = ldpix(x, b, 3, h, w);
    #pragma unroll
    for (int j = 0; j < KNN_; ++j) {
        float dx = pc1[j] - a1, dy = pc2[j] - a2, dz = pc3[j] - a3;
        float xy = dx * dx + dy * dy;
        float r  = sqrtf(xy + dz * dz);
        float th = atan2f(sqrtf(xy), dz * dz);
        float ph = atan2f(dy * dy, dx * dx);
        const int kr = 9 + j, kt = 18 + j, kp = 27 + j;
        pkB[kr >> 3][kr & 7] = (_Float16)r;
        pkB[kt >> 3][kt & 7] = (_Float16)th;
        pkB[kp >> 3][kp & 7] = (_Float16)ph;
    }
    {
        _Float16* fb = &featB[tid * FSTRIDE];
        #pragma unroll
        for (int g = 0; g < 8; ++g) *(v8h*)&fb[g * 8] = pkB[g];
    }

    __syncthreads();

    // ---------------- phase 2: WMMA 45->32 GEMM per branch ----------------
    for (int br = 0; br < 2; ++br) {
        const _Float16* feat = br ? featB : featA;
        const float*    wgt  = br ? wp : wr;                 // [32][45] row-major
        float* obase = out + (size_t)br * ((size_t)4 * STEM_ * HW_);

        #pragma unroll
        for (int m = 0; m < 2; ++m) {                        // two 16-pixel M tiles per wave
            // A fragments: lanes 0-15 M=lane K{0-7,16-23}; lanes 16-31 M=lane-16 K{8-15,24-31}
            const int row = wave * 32 + m * 16 + (lane & 15);
            const _Float16* fr = &feat[row * FSTRIDE];
            const int khalf = ((lane >> 4) & 1) * 8;
            v16h afr[2];
            #pragma unroll
            for (int c = 0; c < 2; ++c) {
                v8h lo = *(const v8h*)&fr[32 * c + khalf];        // K base..base+7
                v8h hi = *(const v8h*)&fr[32 * c + khalf + 16];   // K base+16..base+23
                afr[c] = __builtin_shufflevector(lo, hi,
                          0, 1, 2, 3, 4, 5, 6, 7, 8, 9, 10, 11, 12, 13, 14, 15);
            }
            #pragma unroll
            for (int t = 0; t < 2; ++t) {                    // two 16-channel N tiles
                const int N  = t * 16 + (lane & 15);
                const int kb = ((lane >> 4) & 1) * 8;
                v8f acc = {};
                #pragma unroll
                for (int c = 0; c < 2; ++c) {
                    // B fragment: lanes 0-15 K{0-7,16-23}; lanes 16-31 K{8-15,24-31} (matches A)
                    v16h bfr;
                    #pragma unroll
                    for (int d = 0; d < 8; ++d) {
                        int k0 = 32 * c + kb + 2 * d + (d >> 2) * 8;
                        float w0 = (k0     < 45) ? wgt[N * 45 + k0]     : 0.0f;
                        float w1 = (k0 + 1 < 45) ? wgt[N * 45 + k0 + 1] : 0.0f;
                        bfr[2 * d]     = (_Float16)w0;
                        bfr[2 * d + 1] = (_Float16)w1;
                    }
                    acc = __builtin_amdgcn_wmma_f32_16x16x32_f16(
                        false, afr[c], false, bfr, (short)0, acc, false, false);
                }
                // ReLU + stage (D layout: VGPR i -> M = i + (lane>=16 ? 8 : 0), N = lane%16)
                const int pixb = wave * 32 + m * 16 + ((lane >> 4) & 1) * 8;
                float4 s0 = make_float4(fmaxf(acc[0], 0.0f), fmaxf(acc[1], 0.0f),
                                        fmaxf(acc[2], 0.0f), fmaxf(acc[3], 0.0f));
                float4 s1 = make_float4(fmaxf(acc[4], 0.0f), fmaxf(acc[5], 0.0f),
                                        fmaxf(acc[6], 0.0f), fmaxf(acc[7], 0.0f));
                *(float4*)&ostage[N * OSTRIDE + pixb]     = s0;   // banks 4n..4n+3: conflict-free
                *(float4*)&ostage[N * OSTRIDE + pixb + 4] = s1;
            }
        }
        __syncthreads();
        // coalesced write-out: each thread moves 8 float4 chunks of one channel row
        {
            const int o  = tid >> 3;          // 0..31: output channel
            const int c8 = tid & 7;
            float* grow = obase + ((size_t)b * STEM_ + o) * HW_ + hwBase;
            const float* srow = &ostage[o * OSTRIDE];
            #pragma unroll
            for (int it = 0; it < 8; ++it) {
                int col = (c8 + it * 8) * 4;
                *(float4*)&grow[col] = *(const float4*)&srow[col];  // b128 ds->global
            }
        }
        __syncthreads();
    }
}

extern "C" void kernel_launch(void* const* d_in, const int* in_sizes, int n_in,
                              void* d_out, int out_size, void* d_ws, size_t ws_size,
                              hipStream_t stream) {
    const float* x  = (const float*)d_in[0];
    const float* px = (const float*)d_in[1];
    const float* wr = (const float*)d_in[2];
    const float* wp = (const float*)d_in[3];
    float* out = (float*)d_out;

    dim3 block(256);
    dim3 grid((4 * HW_) / 256);   // 2048 blocks
    knn_conv_kernel<<<grid, block, 0, stream>>>(x, px, wr, wp, out);
}
